// MultiGraphAttention_5377299055202
// MI455X (gfx1250) — compile-verified, matrix operand order checked by reading, not run on hardware
//
#include <hip/hip_runtime.h>

#define BB 4
#define NN 1024
#define DD 512
#define GG 3
#define HH 8
#define DKK 64
#define BN (BB * NN)    // 4096
#define GD3 (GG * DD)   // 1536

typedef __attribute__((ext_vector_type(16))) __bf16 v16bf;
typedef __attribute__((ext_vector_type(8)))  float  v8f;

struct U32x8 { uint4 lo; uint4 hi; };
union FragU { U32x8 u; v16bf v; };

static __device__ __forceinline__ unsigned short f2bf(float f) {
  unsigned int u = __float_as_uint(f);
  u += 0x7FFFu + ((u >> 16) & 1u);      // round-to-nearest-even
  return (unsigned short)(u >> 16);
}

// Load a 16x32 bf16 fragment (A-layout == B^T-layout per CDNA5 ISA 7.12.2):
// lane L: row = L&15, K-base = (L>>4)*8; elems 0..7 = K[kb..kb+7], 8..15 = K[kb+16..kb+23].
static __device__ __forceinline__ v16bf load_frag(const unsigned short* base, int stride, int lane) {
  const int r  = lane & 15;
  const int kb = (lane >> 4) << 3;
  const unsigned short* p = base + (size_t)r * stride + kb;
  FragU c;
  c.u.lo = *(const uint4*)(p);
  c.u.hi = *(const uint4*)(p + 16);
  return c.v;
}

static __device__ __forceinline__ v8f wmma_bf16(v16bf a, v16bf b, v8f c) {
  return __builtin_amdgcn_wmma_f32_16x16x32_bf16(false, a, false, b, (short)0, c, false, false);
}

// ---------------- conversion kernels ----------------

__global__ void k_cvt_x(const float* __restrict__ x, unsigned short* __restrict__ xb, int n) {
  int i = blockIdx.x * 256 + threadIdx.x;
  if (i < n) xb[i] = f2bf(x[i]);
}

// W[g][d][o] fp32 -> wT[(g*3+which)][o][d] bf16  (exactly 9*512*512 threads)
__global__ void k_cvt_wqkv(const float* __restrict__ Wq, const float* __restrict__ Wk,
                           const float* __restrict__ Wv, unsigned short* __restrict__ wT) {
  int i = blockIdx.x * 256 + threadIdx.x;
  int slot = i >> 18;           // / (512*512)
  int rem  = i & 262143;
  int o = rem >> 9, d = rem & 511;
  int g = slot / 3, which = slot % 3;
  const float* W = (which == 0) ? Wq : (which == 1) ? Wk : Wv;
  wT[i] = f2bf(W[((size_t)g * DD + d) * DD + o]);
}

// Wo[i][o] fp32 ([1536][512]) -> woT[o][i] bf16 ([512][1536])
__global__ void k_cvt_wo(const float* __restrict__ Wo, unsigned short* __restrict__ woT) {
  int i = blockIdx.x * 256 + threadIdx.x;   // 512*1536 threads
  int o = i / GD3, c = i % GD3;
  woT[(size_t)o * GD3 + c] = f2bf(Wo[(size_t)c * DD + o]);
}

// ---------------- QKV projection GEMM ----------------
// grid: (BN/128, DD/64, 9)   block: 256 (8 waves, 4x2), 128x64 tile, K=512
__global__ void __launch_bounds__(256) k_gemm_qkv(
    const unsigned short* __restrict__ xb, const unsigned short* __restrict__ wqkvT,
    const float* __restrict__ bq, const float* __restrict__ bk, const float* __restrict__ bv,
    unsigned short* __restrict__ qb, unsigned short* __restrict__ kb2,
    unsigned short* __restrict__ vT) {
  __shared__ __attribute__((aligned(16))) unsigned short At[128][32];
  __shared__ __attribute__((aligned(16))) unsigned short Bt[64][32];
  const int tid = threadIdx.x, lane = tid & 31, w = tid >> 5;
  const int m0 = blockIdx.x * 128, n0 = blockIdx.y * 64;
  const int idx = blockIdx.z;               // g*3 + which
  const int g = idx / 3, which = idx % 3;
  const unsigned short* wt = wqkvT + (size_t)idx * DD * DD;
  const int wm = w & 3, wn = w >> 2;
  v8f acc[2][2] = {};
  for (int k0 = 0; k0 < DD; k0 += 32) {
    __syncthreads();
#pragma unroll
    for (int i = 0; i < 2; ++i) {           // A: 128x32 = 512 16B chunks
      int c = tid * 2 + i;
      int row = c >> 2, coff = (c & 3) * 8;
      *(uint4*)&At[row][coff] = *(const uint4*)(xb + (size_t)(m0 + row) * DD + k0 + coff);
    }
    {                                       // B: 64x32 = 256 16B chunks
      int row = tid >> 2, coff = (tid & 3) * 8;
      *(uint4*)&Bt[row][coff] = *(const uint4*)(wt + (size_t)(n0 + row) * DD + k0 + coff);
    }
    __syncthreads();
#pragma unroll
    for (int fm = 0; fm < 2; ++fm) {
      v16bf a = load_frag(&At[wm * 32 + fm * 16][0], 32, lane);
#pragma unroll
      for (int fn = 0; fn < 2; ++fn) {
        v16bf b = load_frag(&Bt[wn * 32 + fn * 16][0], 32, lane);
        acc[fm][fn] = wmma_bf16(a, b, acc[fm][fn]);
      }
    }
  }
  const float* bias = ((which == 0) ? bq : (which == 1) ? bk : bv) + g * DD;
  const int mo = (lane >> 4) * 8, nl = lane & 15;
#pragma unroll
  for (int fm = 0; fm < 2; ++fm) {
#pragma unroll
    for (int fn = 0; fn < 2; ++fn) {
      int ncol = n0 + wn * 32 + fn * 16 + nl;
      float bsv = bias[ncol];
      int mrow = m0 + wm * 32 + fm * 16 + mo;
#pragma unroll
      for (int j = 0; j < 8; ++j) {
        unsigned short hv = f2bf(acc[fm][fn][j] + bsv);
        if (which == 2)      vT[((size_t)g * DD + ncol) * BN + (mrow + j)] = hv;
        else if (which == 0) qb[((size_t)g * BN + mrow + j) * DD + ncol] = hv;
        else                 kb2[((size_t)g * BN + mrow + j) * DD + ncol] = hv;
      }
    }
  }
}

// ---------------- attention (flash-style online softmax) ----------------
// grid: (NN/128, HH, GG*BB)  block: 256 (8 waves, 16 query rows each)
__global__ void __launch_bounds__(256) k_attn(
    const unsigned short* __restrict__ qb, const unsigned short* __restrict__ kb2,
    const unsigned short* __restrict__ vT, const int* __restrict__ adj,
    unsigned short* __restrict__ comb) {
  __shared__ __attribute__((aligned(16))) unsigned short Pbuf[8][16][32];
  const int tid = threadIdx.x, lane = tid & 31, w = tid >> 5;
  const int h = blockIdx.y;
  const int g = blockIdx.z >> 2, b = blockIdx.z & 3;
  const int rowbase = blockIdx.x * 128 + w * 16;
  const int mo = (lane >> 4) * 8, nl = lane & 15;
  const unsigned short* qbase = qb + ((size_t)g * BN + b * NN + rowbase) * DD + h * DKK;
  const unsigned short* kbase = kb2 + ((size_t)g * BN + b * NN) * DD + h * DKK;
  const unsigned short* vbase = vT + ((size_t)g * DD + h * DKK) * BN + b * NN;
  const int* adjp = adj + (size_t)(g * BB + b) * NN * NN;
  v16bf aq0 = load_frag(qbase, DD, lane);
  v16bf aq1 = load_frag(qbase + 32, DD, lane);
  float mrow[8], lrow[8];
#pragma unroll
  for (int j = 0; j < 8; ++j) { mrow[j] = -3.0e38f; lrow[j] = 0.0f; }
  v8f acc[4] = {};
  for (int kc = 0; kc < NN / 32; ++kc) {
    __builtin_prefetch(kbase + (size_t)(kc * 32 + 32) * DD, 0, 0);
#pragma unroll
    for (int sub = 0; sub < 2; ++sub) {
      const int col0 = kc * 32 + sub * 16;
      v16bf bk0 = load_frag(kbase + (size_t)col0 * DD, DD, lane);
      v16bf bk1 = load_frag(kbase + (size_t)col0 * DD + 32, DD, lane);
      v8f s = {};
      s = wmma_bf16(aq0, bk0, s);
      s = wmma_bf16(aq1, bk1, s);
      const int ncol = col0 + nl;
      float sc[8];
#pragma unroll
      for (int j = 0; j < 8; ++j) {
        float sv = s[j] * 0.125f;           // 1/sqrt(DK)
        int am = adjp[(size_t)(rowbase + mo + j) * NN + ncol];
        sv = (am == 0) ? -1.0e9f : sv;
        float t = sv;                        // row max over 16 lanes
        t = fmaxf(t, __shfl_xor(t, 1, 16));
        t = fmaxf(t, __shfl_xor(t, 2, 16));
        t = fmaxf(t, __shfl_xor(t, 4, 16));
        t = fmaxf(t, __shfl_xor(t, 8, 16));
        float mn = fmaxf(mrow[j], t);
        float scale = __expf(mrow[j] - mn);
        float p = __expf(sv - mn);
        float ps = p;                        // row sum over 16 lanes
        ps += __shfl_xor(ps, 1, 16);
        ps += __shfl_xor(ps, 2, 16);
        ps += __shfl_xor(ps, 4, 16);
        ps += __shfl_xor(ps, 8, 16);
        lrow[j] = lrow[j] * scale + ps;
        mrow[j] = mn;
        sc[j] = scale;
        Pbuf[w][mo + j][sub * 16 + nl] = f2bf(p);
      }
#pragma unroll
      for (int f = 0; f < 4; ++f)
#pragma unroll
        for (int j = 0; j < 8; ++j) acc[f][j] *= sc[j];
    }
    asm volatile("s_wait_dscnt 0" ::: "memory");   // per-wave Pbuf write -> read
    v16bf ap = load_frag(&Pbuf[w][0][0], 32, lane);
#pragma unroll
    for (int f = 0; f < 4; ++f) {
      v16bf bvv = load_frag(vbase + (size_t)(f * 16) * BN + kc * 32, BN, lane);
      acc[f] = wmma_bf16(ap, bvv, acc[f]);
    }
  }
  unsigned short* cb = comb + (size_t)(b * NN + rowbase) * GD3 + g * DD + h * DKK;
#pragma unroll
  for (int j = 0; j < 8; ++j) {
    float inv = 1.0f / lrow[j];
#pragma unroll
    for (int f = 0; f < 4; ++f)
      cb[(size_t)(mo + j) * GD3 + f * 16 + nl] = f2bf(acc[f][j] * inv);
  }
}

// ---------------- output projection + residual + LayerNorm ----------------
// grid: BN/16  block: 256 (8 waves x 64 cols), 16 rows x 512 cols per block, K=1536
__global__ void __launch_bounds__(256) k_out_ln(
    const unsigned short* __restrict__ comb, const unsigned short* __restrict__ woT,
    const float* __restrict__ bo, const float* __restrict__ x,
    const float* __restrict__ gamma, const float* __restrict__ beta,
    float* __restrict__ out) {
  __shared__ __attribute__((aligned(16))) unsigned short At[16][32];
  __shared__ float red[8][16][2];
  const int tid = threadIdx.x, lane = tid & 31, w = tid >> 5;
  const int row0 = blockIdx.x * 16;
  const int mo = (lane >> 4) * 8, nl = lane & 15;
  v8f acc[4] = {};
  for (int k0 = 0; k0 < GD3; k0 += 32) {
    __syncthreads();
    if (tid < 64) {
      int row = tid >> 2, coff = (tid & 3) * 8;
      *(uint4*)&At[row][coff] = *(const uint4*)(comb + (size_t)(row0 + row) * GD3 + k0 + coff);
    }
    __syncthreads();
    v16bf a = load_frag(&At[0][0], 32, lane);
#pragma unroll
    for (int f = 0; f < 4; ++f) {
      v16bf bw = load_frag(woT + (size_t)(w * 64 + f * 16) * GD3 + k0, GD3, lane);
      acc[f] = wmma_bf16(a, bw, acc[f]);
    }
  }
  float ps[8], ss[8];
#pragma unroll
  for (int j = 0; j < 8; ++j) { ps[j] = 0.f; ss[j] = 0.f; }
#pragma unroll
  for (int f = 0; f < 4; ++f) {
    int ncol = w * 64 + f * 16 + nl;
    float bof = bo[ncol];
#pragma unroll
    for (int j = 0; j < 8; ++j) {
      float z = acc[f][j] + bof + x[(size_t)(row0 + mo + j) * DD + ncol];
      acc[f][j] = z;
      ps[j] += z; ss[j] += z * z;
    }
  }
#pragma unroll
  for (int j = 0; j < 8; ++j) {
    float a1 = ps[j], a2 = ss[j];
    a1 += __shfl_xor(a1, 1, 16); a2 += __shfl_xor(a2, 1, 16);
    a1 += __shfl_xor(a1, 2, 16); a2 += __shfl_xor(a2, 2, 16);
    a1 += __shfl_xor(a1, 4, 16); a2 += __shfl_xor(a2, 4, 16);
    a1 += __shfl_xor(a1, 8, 16); a2 += __shfl_xor(a2, 8, 16);
    ps[j] = a1; ss[j] = a2;
  }
  if (nl == 0) {
#pragma unroll
    for (int j = 0; j < 8; ++j) { red[w][mo + j][0] = ps[j]; red[w][mo + j][1] = ss[j]; }
  }
  __syncthreads();
#pragma unroll
  for (int j = 0; j < 8; ++j) {
    float tot = 0.f, tsq = 0.f;
#pragma unroll
    for (int w2 = 0; w2 < 8; ++w2) { tot += red[w2][mo + j][0]; tsq += red[w2][mo + j][1]; }
    float mu = tot * (1.0f / DD);
    float var = tsq * (1.0f / DD) - mu * mu;
    float rs = rsqrtf(var + 1e-5f);
#pragma unroll
    for (int f = 0; f < 4; ++f) {
      int ncol = w * 64 + f * 16 + nl;
      out[(size_t)(row0 + mo + j) * DD + ncol] = (acc[f][j] - mu) * rs * gamma[ncol] + beta[ncol];
    }
  }
}

extern "C" void kernel_launch(void* const* d_in, const int* in_sizes, int n_in,
                              void* d_out, int out_size, void* d_ws, size_t ws_size,
                              hipStream_t stream) {
  (void)in_sizes; (void)n_in; (void)out_size; (void)ws_size;
  const float* x     = (const float*)d_in[0];
  const int*   adj   = (const int*)d_in[1];
  const float* Wq    = (const float*)d_in[2];
  const float* bq    = (const float*)d_in[3];
  const float* Wk    = (const float*)d_in[4];
  const float* bk    = (const float*)d_in[5];
  const float* Wv    = (const float*)d_in[6];
  const float* bv    = (const float*)d_in[7];
  const float* Wo    = (const float*)d_in[8];
  const float* bo    = (const float*)d_in[9];
  const float* gamma = (const float*)d_in[10];
  const float* beta  = (const float*)d_in[11];
  float* out = (float*)d_out;

  unsigned short* p = (unsigned short*)d_ws;
  unsigned short* xb    = p; p += (size_t)BN * DD;        // 2.10M elems
  unsigned short* wqkvT = p; p += (size_t)9 * DD * DD;    // 2.36M
  unsigned short* woT   = p; p += (size_t)DD * GD3;       // 0.79M
  unsigned short* qb    = p; p += (size_t)GG * BN * DD;   // 6.29M
  unsigned short* kb2   = p; p += (size_t)GG * BN * DD;   // 6.29M
  unsigned short* vT    = p; p += (size_t)GG * DD * BN;   // 6.29M
  unsigned short* comb  = p; p += (size_t)BN * GD3;       // 6.29M  (~61 MB total)

  k_cvt_x<<<(BN * DD) / 256, 256, 0, stream>>>(x, xb, BN * DD);
  k_cvt_wqkv<<<(9 * DD * DD) / 256, 256, 0, stream>>>(Wq, Wk, Wv, wqkvT);
  k_cvt_wo<<<(DD * GD3) / 256, 256, 0, stream>>>(Wo, woT);
  k_gemm_qkv<<<dim3(BN / 128, DD / 64, 9), 256, 0, stream>>>(xb, wqkvT, bq, bk, bv, qb, kb2, vT);
  k_attn<<<dim3(NN / 128, HH, GG * BB), 256, 0, stream>>>(qb, kb2, vT, adj, comb);
  k_out_ln<<<BN / 16, 256, 0, stream>>>(comb, woT, bo, x, gamma, beta, out);
}
